// HierarchicalPrototypeClassifier_3049426780612
// MI455X (gfx1250) — matrix operand order repaired and checked by reading, not compiled, not run on hardware
//
#include <hip/hip_runtime.h>
#include <hip/hip_bf16.h>
#include <math.h>

// ---------------- problem constants ----------------
#define BB 4096
#define CC 1000
#define PP 10
#define DD 256
#define NN 10000      // CC*PP prototypes
#define EE 160000

typedef _Float16 half8  __attribute__((ext_vector_type(8)));
typedef _Float16 v16h   __attribute__((ext_vector_type(16)));
typedef float    v8f    __attribute__((ext_vector_type(8)));

// ---- WMMA fragment loaders (v_wmma_f32_16x16x32_f16; contiguous k order, same for A and B) ----
// A: 16x32 f16. lane l: m = l&15, hi = l>>4; elem e -> K=k0+hi*16+e.
// Row-major source, stride in halves.
__device__ __forceinline__ v16h load_a_frag(const _Float16* A, int row0, int stride,
                                            int k0, int lane) {
  int m = lane & 15, hi = lane >> 4;
  const _Float16* p = A + (size_t)(row0 + m) * stride + k0 + hi * 16;
  half8 lo = *(const half8*)p;
  half8 hb = *(const half8*)(p + 8);
  v16h r;
#pragma unroll
  for (int i = 0; i < 8; ++i) { r[i] = lo[i]; r[8 + i] = hb[i]; }
  return r;
}

// B: 32x16 f16, B[k][n] = M[n][k] (i.e. multiply by M^T). lane l: n = l&15,
// elem e -> K = k0 + (l>>4)*16 + e : 16 contiguous halves of row n of M.
__device__ __forceinline__ v16h load_bT_frag(const _Float16* M, int col0, int stride,
                                             int k0, int lane) {
  int n = lane & 15, hi = lane >> 4;
  return *(const v16h*)(M + (size_t)(col0 + n) * stride + k0 + hi * 16);
}

// B from pre-packed weight buffer: [(kt*16+ct)*32+lane]*16 contiguous halves.
__device__ __forceinline__ v16h load_b_packed(const _Float16* Wp, int kt, int ct, int lane) {
  return *(const v16h*)(Wp + (((size_t)(kt * 16 + ct) * 32 + lane) << 4));
}

// ---------------- init ----------------
__global__ void k_init(float* pos, float* neg, float* svec, float* cecnt, float* deg) {
  int i = blockIdx.x * 256 + threadIdx.x;
  if (i < NN) { pos[i] = 0.f; neg[i] = 0.f; deg[i] = 1.0f; }   // deg=1: self-loop
  if (i < DD) svec[i] = 0.f;
  if (i < 2)  cecnt[i] = 0.f;
}

__global__ void k_deg_edges(const int* dst, float* deg) {
  int i = blockIdx.x * 256 + threadIdx.x;
  if (i < EE) atomicAdd(&deg[dst[i]], 1.0f);
}

__global__ void k_deg_rsqrt(float* deg) {
  int i = blockIdx.x * 256 + threadIdx.x;
  if (i < NN) deg[i] = rsqrtf(deg[i]);          // deg >= 1 always
}

__global__ void k_cvt_f16(const float* x, _Float16* xh) {
  size_t i = (size_t)blockIdx.x * 256 + threadIdx.x;   // N*D threads
  xh[i] = (_Float16)x[i];
}

// pack w[k][n] (row-major DxD f32) into WMMA-B fragment order, f16
__global__ void k_pack_w(const float* w, _Float16* wp) {
  int idx = blockIdx.x * 256 + threadIdx.x;            // D*D = 65536
  int e = idx & 15, lane = (idx >> 4) & 31, ct = (idx >> 9) & 15, kt = idx >> 13;
  int k = kt * 32 + (lane >> 4) * 16 + e;
  int n = ct * 16 + (lane & 15);
  wp[idx] = (_Float16)w[(size_t)k * DD + n];
}

// ------- GCN dense GEMM: xw = Ah(NNxD) @ Wpacked(DxD) -------
// One wave computes 16 rows x 64 cols (4 accumulators share one A fragment).
__global__ void __launch_bounds__(32) k_gemm_xw(const _Float16* Ah, const _Float16* Wp,
                                                float* Co) {
  int lane = threadIdx.x;
  int rt = blockIdx.x;           // 0..624 (16-row tiles)
  int cq = blockIdx.y;           // 0..3   (64-col quadrants)
  v8f c[4] = {};
#pragma unroll
  for (int kt = 0; kt < 8; ++kt) {
    v16h a = load_a_frag(Ah, rt * 16, DD, kt * 32, lane);
#pragma unroll
    for (int t = 0; t < 4; ++t) {
      v16h b = load_b_packed(Wp, kt, cq * 4 + t, lane);
      c[t] = __builtin_amdgcn_wmma_f32_16x16x32_f16(false, a, false, b, (short)0,
                                                    c[t], false, false);
    }
  }
  int m0 = (lane >> 4) * 8, nn = lane & 15;
#pragma unroll
  for (int t = 0; t < 4; ++t)
#pragma unroll
    for (int r = 0; r < 8; ++r)
      Co[(size_t)(rt * 16 + m0 + r) * DD + (cq * 4 + t) * 16 + nn] = c[t][r];
}

// agg[n][d] = dii[n]^2 * xw[n][d]   (self-loop term; also zero-initializes agg)
__global__ void k_selfloop(const float* xw, const float* dii, float* agg) {
  size_t i = (size_t)blockIdx.x * 256 + threadIdx.x;   // N*D
  int n = (int)(i >> 8);
  float w = dii[n];
  agg[i] = w * w * xw[i];
}

// per-edge scatter: agg[dst] += dii[src]*dii[dst] * xw[src]
__global__ void __launch_bounds__(256) k_edges(const int* src, const int* dst,
                                               const float* dii, const float* xw,
                                               float* agg) {
  int e = blockIdx.x;
  int d = threadIdx.x;
  int s = src[e], t = dst[e];
  float w = dii[s] * dii[t];
  atomicAdd(&agg[(size_t)t * DD + d], w * xw[(size_t)s * DD + d]);
}

// xh = f16(relu(agg + b))
__global__ void k_bias_relu_f16(const float* agg, const float* b, _Float16* xh) {
  size_t i = (size_t)blockIdx.x * 256 + threadIdx.x;
  float v = agg[i] + b[(int)(i & 255)];
  xh[i] = (_Float16)(v > 0.f ? v : 0.f);
}

// layer-2 epilogue: bias + relu + row-normalize -> f16, plus dispersion partial sums
__global__ void __launch_bounds__(256) k_norm_proto(const float* agg, const float* b2,
                                                    _Float16* ph, float* svec) {
  int n = blockIdx.x, d = threadIdx.x;
  float v = agg[(size_t)n * DD + d] + b2[d];
  v = v > 0.f ? v : 0.f;
  __shared__ float red[256];
  red[d] = v * v; __syncthreads();
  for (int s = 128; s > 0; s >>= 1) { if (d < s) red[d] += red[d + s]; __syncthreads(); }
  float nrm = fmaxf(sqrtf(red[0]), 1e-12f);
  float p = v / nrm;
  ph[(size_t)n * DD + d] = (_Float16)p;
  atomicAdd(&svec[d], p);
}

__global__ void __launch_bounds__(256) k_norm_hidden(const float* h, _Float16* hh) {
  int n = blockIdx.x, d = threadIdx.x;
  float v = h[(size_t)n * DD + d];
  __shared__ float red[256];
  red[d] = v * v; __syncthreads();
  for (int s = 128; s > 0; s >>= 1) { if (d < s) red[d] += red[d + s]; __syncthreads(); }
  float nrm = fmaxf(sqrtf(red[0]), 1e-12f);
  hh[(size_t)n * DD + d] = (_Float16)(v / nrm);
}

// ---------------- fused sim GEMM + per-class max/argmax ----------------
// Block: 128 rows x 80 cols (= 8 classes exactly). 4 waves; wave w owns rows
// [32w, 32w+32): two A fragments amortized over 5 shared B column tiles
// -> 10 independent accumulators per wave (no WMMA RAW bubbles).
__global__ void __launch_bounds__(128) k_sim(const _Float16* Hh, const _Float16* Ph,
                                             float* out, int* topk) {
  __shared__ float ls[128 * 80];                 // 40 KB
  int lane = threadIdx.x & 31;
  int wave = threadIdx.x >> 5;
  int ctile = blockIdx.x;           // 0..124 -> cols 80*ctile, classes 8*ctile
  int rtile = blockIdx.y;           // 0..31  -> rows 128*rtile
  int row0 = rtile * 128 + wave * 32;
  int col0 = ctile * 80;

  v8f c[2][5] = {};
  for (int kt = 0; kt < 8; ++kt) {
    v16h a0 = load_a_frag(Hh, row0,      DD, kt * 32, lane);
    v16h a1 = load_a_frag(Hh, row0 + 16, DD, kt * 32, lane);
#pragma unroll
    for (int t = 0; t < 5; ++t) {
      v16h b = load_bT_frag(Ph, col0 + t * 16, DD, kt * 32, lane);
      c[0][t] = __builtin_amdgcn_wmma_f32_16x16x32_f16(false, a0, false, b, (short)0,
                                                       c[0][t], false, false);
      c[1][t] = __builtin_amdgcn_wmma_f32_16x16x32_f16(false, a1, false, b, (short)0,
                                                       c[1][t], false, false);
    }
  }
  int m0 = (lane >> 4) * 8, nn = lane & 15;
#pragma unroll
  for (int h = 0; h < 2; ++h)
#pragma unroll
    for (int t = 0; t < 5; ++t)
#pragma unroll
      for (int r = 0; r < 8; ++r)
        ls[(wave * 32 + h * 16 + m0 + r) * 80 + t * 16 + nn] = c[h][t][r];
  __syncthreads();

  // 128 rows x 8 classes = 1024 (row,class) reductions over P=10
  for (int pair = threadIdx.x; pair < 1024; pair += 128) {
    int r = pair >> 3, c8 = pair & 7;
    const float* p = &ls[r * 80 + c8 * 10];
    float best = p[0]; int arg = 0;
#pragma unroll
    for (int j = 1; j < 10; ++j) { float v = p[j]; if (v > best) { best = v; arg = j; } }
    int b = rtile * 128 + r, cls = ctile * 8 + c8;
    out[(size_t)b * CC + cls] = 0.5f * (1.0f + best);
    topk[(size_t)b * CC + cls] = arg;
  }
}

// ---------------- per-row loss + usage scatters ----------------
__global__ void __launch_bounds__(256) k_loss(const float* out, const int* labels,
                                              const int* topk, float* pos, float* neg,
                                              float* cecnt) {
  int b = blockIdx.x;
  const float* row = out + (size_t)b * CC;
  __shared__ float smax[256]; __shared__ int sarg[256]; __shared__ float ssum[256];
  float m = -1e30f; int a = 0;
  for (int c = threadIdx.x; c < CC; c += 256) {
    float v = row[c];
    if (v > m) { m = v; a = c; }
  }
  smax[threadIdx.x] = m; sarg[threadIdx.x] = a; __syncthreads();
  for (int s = 128; s > 0; s >>= 1) {
    if (threadIdx.x < s) {
      float vo = smax[threadIdx.x + s]; int ao = sarg[threadIdx.x + s];
      if (vo > smax[threadIdx.x] ||
          (vo == smax[threadIdx.x] && ao < sarg[threadIdx.x])) {
        smax[threadIdx.x] = vo; sarg[threadIdx.x] = ao;
      }
    }
    __syncthreads();
  }
  float gmax = smax[0]; int pred = sarg[0];
  float se = 0.f;
  for (int c = threadIdx.x; c < CC; c += 256) se += __expf(row[c] - gmax);
  ssum[threadIdx.x] = se; __syncthreads();
  for (int s = 128; s > 0; s >>= 1) {
    if (threadIdx.x < s) ssum[threadIdx.x] += ssum[threadIdx.x + s];
    __syncthreads();
  }
  if (threadIdx.x == 0) {
    int l = labels[b];
    float ts = row[l];
    if ((ts - gmax) > -0.1f) {                      // margin mask
      float ce = -(ts - gmax - __logf(ssum[0]));    // -log_softmax[label]
      atomicAdd(&cecnt[0], ce);
      atomicAdd(&cecnt[1], 1.0f);
    }
    atomicAdd(&pos[l * PP + topk[(size_t)b * CC + l]], 1.0f);
    if (pred != l) atomicAdd(&neg[pred * PP + topk[(size_t)b * CC + pred]], 1.0f);
  }
}

__global__ void __launch_bounds__(256) k_fin(const float* svec, const float* cecnt,
                                             float* loss_out) {
  __shared__ float red[256];
  float v = svec[threadIdx.x];
  red[threadIdx.x] = v * v; __syncthreads();
  for (int s = 128; s > 0; s >>= 1) {
    if (threadIdx.x < s) red[threadIdx.x] += red[threadIdx.x + s];
    __syncthreads();
  }
  if (threadIdx.x == 0) {
    float cnt = cecnt[1];
    float disp = -red[0] / ((float)NN * (float)NN);
    *loss_out = (cnt == 0.f) ? 0.f : (cecnt[0] / fmaxf(cnt, 1.f) + 0.1f * disp);
  }
}

// ---------------- launch ----------------
extern "C" void kernel_launch(void* const* d_in, const int* in_sizes, int n_in,
                              void* d_out, int out_size, void* d_ws, size_t ws_size,
                              hipStream_t stream) {
  const float* hidden = (const float*)d_in[0];
  const int*   labels = (const int*)d_in[1];
  const int*   edge   = (const int*)d_in[2];
  const float* x      = (const float*)d_in[3];
  const float* w1     = (const float*)d_in[4];
  const float* b1     = (const float*)d_in[5];
  const float* w2     = (const float*)d_in[6];
  const float* b2     = (const float*)d_in[7];
  const int* src = edge;
  const int* dst = edge + EE;

  char* ws = (char*)d_ws;
  size_t off = 0;
  auto alloc = [&](size_t bytes) -> char* {
    off = (off + 255) & ~(size_t)255;
    char* p = ws + off; off += bytes; return p;
  };
  float*     deg   = (float*)alloc((size_t)NN * 4);
  _Float16*  xh    = (_Float16*)alloc((size_t)NN * DD * 2);
  _Float16*  wpack = (_Float16*)alloc((size_t)DD * DD * 2);
  float*     xw    = (float*)alloc((size_t)NN * DD * 4);
  float*     agg   = (float*)alloc((size_t)NN * DD * 4);
  _Float16*  hidh  = (_Float16*)alloc((size_t)BB * DD * 2);
  _Float16*  ph    = (_Float16*)alloc((size_t)NN * DD * 2);
  int*       topk  = (int*)alloc((size_t)BB * CC * 4);
  float*     svec  = (float*)alloc((size_t)DD * 4);
  float*     cecnt = (float*)alloc(2 * 4);

  float* out      = (float*)d_out;                 // [B,C]
  float* out_loss = out + (size_t)BB * CC;         // [1]
  float* pos      = out_loss + 1;                  // [N]
  float* neg      = pos + NN;                      // [N]

  const int ND_BLKS = (NN * DD) / 256;             // 10000

  k_init<<<(NN + 255) / 256, 256, 0, stream>>>(pos, neg, svec, cecnt, deg);
  k_deg_edges<<<(EE + 255) / 256, 256, 0, stream>>>(dst, deg);
  k_deg_rsqrt<<<(NN + 255) / 256, 256, 0, stream>>>(deg);
  k_cvt_f16<<<ND_BLKS, 256, 0, stream>>>(x, xh);

  dim3 gg(NN / 16, 4);                             // (625,4): 16 rows x 64 cols / wave
  // ---- GCN layer 1 ----
  k_pack_w<<<(DD * DD) / 256, 256, 0, stream>>>(w1, wpack);
  k_gemm_xw<<<gg, 32, 0, stream>>>(xh, wpack, xw);
  k_selfloop<<<ND_BLKS, 256, 0, stream>>>(xw, deg, agg);
  k_edges<<<EE, 256, 0, stream>>>(src, dst, deg, xw, agg);
  k_bias_relu_f16<<<ND_BLKS, 256, 0, stream>>>(agg, b1, xh);
  // ---- GCN layer 2 ----
  k_pack_w<<<(DD * DD) / 256, 256, 0, stream>>>(w2, wpack);
  k_gemm_xw<<<gg, 32, 0, stream>>>(xh, wpack, xw);
  k_selfloop<<<ND_BLKS, 256, 0, stream>>>(xw, deg, agg);
  k_edges<<<EE, 256, 0, stream>>>(src, dst, deg, xw, agg);
  // ---- normalize ----
  k_norm_proto<<<NN, 256, 0, stream>>>(agg, b2, ph, svec);
  k_norm_hidden<<<BB, 256, 0, stream>>>(hidden, hidh);
  // ---- fused sim GEMM + per-class max ----
  dim3 gs(CC / 8, BB / 128);                       // (125,32)
  k_sim<<<gs, 128, 0, stream>>>(hidh, ph, out, topk);
  // ---- loss + usage + finalize ----
  k_loss<<<BB, 256, 0, stream>>>(out, labels, topk, pos, neg, cecnt);
  k_fin<<<1, 256, 0, stream>>>(svec, cecnt, out_loss);
}
